// Qwen2DecoderLayer_2293512536852
// MI455X (gfx1250) — matrix operand order repaired and checked by reading, not compiled, not run on hardware
//
#include <hip/hip_runtime.h>
#include <hip/hip_bf16.h>
#include <stdint.h>

typedef __bf16 bf16;
typedef __attribute__((ext_vector_type(16))) __bf16 v16bf;
typedef __attribute__((ext_vector_type(8)))  __bf16 bf16x8;
typedef __attribute__((ext_vector_type(8)))  float   v8f;
typedef __attribute__((ext_vector_type(4)))  unsigned int u32x4;
typedef __attribute__((ext_vector_type(8)))  int i32x8;
typedef __attribute__((ext_vector_type(4)))  int i32x4;

#define TOK    4096
#define HIDDEN 2048
#define NHEAD  16
#define NKVH   4
#define HEADD  128
#define FFDIM  8192
#define SEQ    1024
#define BATCH  4

#if defined(__has_builtin)
#if __has_builtin(__builtin_amdgcn_tensor_load_to_lds)
#define HAVE_TDM 1
#endif
#endif

__device__ __forceinline__ bf16 f2bf(float f) {
  unsigned u = __builtin_bit_cast(unsigned, f);
  unsigned r = u + 0x7FFFu + ((u >> 16) & 1u);           // round-to-nearest-even
  unsigned short h = (unsigned short)(r >> 16);
  return __builtin_bit_cast(bf16, h);
}
__device__ __forceinline__ float bf2f(bf16 b) {
  unsigned u = ((unsigned)__builtin_bit_cast(unsigned short, b)) << 16;
  return __builtin_bit_cast(float, u);
}

// LDS byte-offset of a __shared__ object (flat LDS addresses carry the offset in the low 32 bits)
__device__ __forceinline__ unsigned lds_off(const void* p) {
  return (unsigned)(uintptr_t)p;
}

// async global->LDS 16B copy (ASYNCcnt path)
__device__ __forceinline__ void async_copy_b128(unsigned lds_byte_off, const void* gsrc) {
  asm volatile("global_load_async_to_lds_b128 %0, %1, off"
               :: "v"(lds_byte_off), "v"((unsigned long long)(uintptr_t)gsrc)
               : "memory");
}
__device__ __forceinline__ void wait_async0() {
  asm volatile("s_wait_asynccnt 0x0" ::: "memory");
}

#ifdef HAVE_TDM
// TDM: load a 128x128 bf16 tile (row stride = row_stride_elems) into LDS with a
// hardware-padded row stride of 136 bf16 (pad 4 DWORDs every 64 DWORDs).
__device__ __forceinline__ void tdm_load_tile_128x128_bf16(
    unsigned lds_byte_off, const bf16* gptr, unsigned row_stride_elems)
{
  unsigned long long ga = (unsigned long long)(uintptr_t)gptr;
  u32x4 g0;
  g0[0] = 1u;                                             // count=1, user mode
  g0[1] = lds_byte_off;                                   // lds_addr
  g0[2] = (unsigned)(ga & 0xFFFFFFFFu);                   // global_addr[31:0]
  g0[3] = (unsigned)((ga >> 32) & 0x1FFFFFFu) | (2u << 30); // global_addr[56:32] | type=2
  i32x8 g1;
  g1[0] = (int)((1u << 16)      // data_size = 2 bytes
              | (1u << 20)      // pad_enable
              | (5u << 22)      // pad_interval: 64 DWORDs
              | (3u << 25));    // pad_amount:  4 DWORDs  -> LDS stride 272B
  g1[1] = (int)((row_stride_elems & 0xFFFFu) << 16);      // tensor_dim0[15:0]
  g1[2] = (int)(((row_stride_elems >> 16) & 0xFFFFu)      // tensor_dim0[31:16]
              | (4096u << 16));                           // tensor_dim1[15:0]
  g1[3] = (int)(128u << 16);                              // tensor_dim1[31:16]=0 | tile_dim0=128
  g1[4] = (int)128u;                                      // tile_dim1=128, tile_dim2=0
  g1[5] = (int)row_stride_elems;                          // tensor_dim0_stride[31:0]
  g1[6] = 0;                                              // stride hi / dim1_stride lo
  g1[7] = 0;
  i32x4 z4; z4[0] = 0; z4[1] = 0; z4[2] = 0; z4[3] = 0;
  i32x8 z8;
#pragma unroll
  for (int e = 0; e < 8; ++e) z8[e] = 0;
  __builtin_amdgcn_tensor_load_to_lds(g0, g1, z4, z4, z8, 0);   // 6-arg toolchain variant
}
#endif

// ---- WMMA fragment loads from LDS (16-bit A 16x32 / B 32x16 layouts) ----
__device__ __forceinline__ v16bf ld_afrag(const bf16* s, int row, int ld, int kbase, int hs) {
  const bf16* p = s + row * ld + kbase + hs * 8;
  bf16x8 lo = *(const bf16x8*)p;          // K = hs*8 + 0..7
  bf16x8 hi = *(const bf16x8*)(p + 16);   // K = hs*8 + 16..23
  v16bf a;
#pragma unroll
  for (int e = 0; e < 8; ++e) { a[e] = lo[e]; a[e + 8] = hi[e]; }
  return a;
}
__device__ __forceinline__ v16bf ld_bfrag(const bf16* s, int col, int ld, int kbase, int hs) {
  const bf16* p = s + col * ld + kbase + hs * 8;  // K = hs*8 + 0..7, hs*8 + 16..23
  bf16x8 lo = *(const bf16x8*)p;
  bf16x8 hi = *(const bf16x8*)(p + 16);
  v16bf b;
#pragma unroll
  for (int e = 0; e < 8; ++e) { b[e] = lo[e]; b[e + 8] = hi[e]; }
  return b;
}

// =====================  Generic GEMM: C = A(bf16) * W(f32,[N,K])^T (+bias) =====================
#define GBM 128
#define GBN 128
#define GBK 32
#define GLDA 40
#define GLDB 40

template <bool OUT_BF16, bool HAS_BIAS>
__global__ __launch_bounds__(256) void gemm_bf16_kernel(
    const bf16* __restrict__ A, const float* __restrict__ W,
    const float* __restrict__ bias, void* __restrict__ Cout,
    int M, int N, int K)
{
  __shared__ __attribute__((aligned(16))) bf16 sA[GBM * GLDA];
  __shared__ __attribute__((aligned(16))) bf16 sB[GBN * GLDB];
  const int tid = threadIdx.x;
  const int lane = tid & 31, wave = tid >> 5;
  const int wm = wave & 3, wn = wave >> 2;
  const int hs = lane >> 4, lrow = lane & 15;
  const int m0 = blockIdx.x * GBM;
  const int n0 = blockIdx.y * GBN;

  v8f acc[2][4];
#pragma unroll
  for (int i = 0; i < 2; ++i)
#pragma unroll
    for (int j = 0; j < 4; ++j)
#pragma unroll
      for (int v = 0; v < 8; ++v) acc[i][j][v] = 0.0f;

  const int ra = tid >> 1;
  const int ca = (tid & 1) * 16;
  const unsigned ldsA = lds_off(&sA[ra * GLDA + ca]);
  for (int k0 = 0; k0 < K; k0 += GBK) {
    {
      // A tile via async global->LDS (bf16, no conversion needed)
      const bf16* src = A + (size_t)(m0 + ra) * K + k0 + ca;
      async_copy_b128(ldsA, src);
      async_copy_b128(ldsA + 16u, src + 8);
    }
    {
      // W tile through VGPRs with inline f32->bf16 conversion
      const float* src = W + (size_t)(n0 + ra) * K + k0 + ca;
      bf16x8 b0, b1;
#pragma unroll
      for (int e = 0; e < 8; ++e) { b0[e] = f2bf(src[e]); b1[e] = f2bf(src[8 + e]); }
      *(bf16x8*)&sB[ra * GLDB + ca]     = b0;
      *(bf16x8*)&sB[ra * GLDB + ca + 8] = b1;
    }
    wait_async0();
    __syncthreads();
    v16bf af[2], bg[4];
#pragma unroll
    for (int i = 0; i < 2; ++i) af[i] = ld_afrag(sA, wm * 32 + i * 16 + lrow, GLDA, 0, hs);
#pragma unroll
    for (int j = 0; j < 4; ++j) bg[j] = ld_bfrag(sB, wn * 64 + j * 16 + lrow, GLDB, 0, hs);
#pragma unroll
    for (int i = 0; i < 2; ++i)
#pragma unroll
      for (int j = 0; j < 4; ++j)
        acc[i][j] = __builtin_amdgcn_wmma_f32_16x16x32_bf16(
            false, af[i], false, bg[j], (short)0, acc[i][j], false, false);
    __syncthreads();
  }
#pragma unroll
  for (int j = 0; j < 4; ++j) {
    const int col = n0 + wn * 64 + j * 16 + lrow;
    const float bv = HAS_BIAS ? bias[col] : 0.0f;
#pragma unroll
    for (int i = 0; i < 2; ++i) {
      const size_t rbase = (size_t)(m0 + wm * 32 + i * 16 + 8 * hs);
#pragma unroll
      for (int v = 0; v < 8; ++v) {
        const float val = acc[i][j][v] + bv;
        if (OUT_BF16) ((bf16*)Cout)[(rbase + v) * N + col] = f2bf(val);
        else          ((float*)Cout)[(rbase + v) * N + col] = val;
      }
    }
  }
}

// =====================  Flash attention (causal, GQA 16q/4kv, HD=128)  =====================
#define ALD 136          // bf16 LDS row stride (272B; matches TDM pad 64DW+4DW)
#define SLD 132          // f32 score row stride
#define PLD 264          // bf16 stride overlaying the f32 score rows

__global__ __launch_bounds__(256) void attention_kernel(
    const bf16* __restrict__ Q, const bf16* __restrict__ Kmat,
    const bf16* __restrict__ V, bf16* __restrict__ O)
{
  __shared__ __attribute__((aligned(16))) bf16  sQ[128 * ALD];
  __shared__ __attribute__((aligned(16))) bf16  sK[128 * ALD];
  __shared__ __attribute__((aligned(16))) bf16  sV[128 * ALD];   // [hd][key]
  __shared__ __attribute__((aligned(16))) float sS[128 * SLD];   // scores; P(bf16) aliased
  __shared__ float sScale[128];

  bf16* sP = (bf16*)sS;

  const int tid = threadIdx.x;
  const int lane = tid & 31, wave = tid >> 5;
  const int wm = wave & 3, wn = wave >> 2;
  const int hs = lane >> 4, lrow = lane & 15;
  const int qb = blockIdx.x;
  const int head = blockIdx.y;
  const int b = blockIdx.z;
  const int kvh = head >> 2;
  const size_t tok0q = (size_t)b * SEQ + (size_t)qb * 128;
  const float scale = 0.08838834764831845f;   // 1/sqrt(128)

#ifdef HAVE_TDM
  if (tid == 0)      // TDM: Q tile 128x128 bf16 -> LDS with padded stride
    tdm_load_tile_128x128_bf16(lds_off(sQ), Q + tok0q * (NHEAD * HEADD) + head * HEADD,
                               NHEAD * HEADD);
#else
#pragma unroll
  for (int r = 0; r < 8; ++r) {
    int e = tid + r * 256;
    int row = e >> 4, c = (e & 15) * 8;
    const bf16* src = Q + (tok0q + row) * (NHEAD * HEADD) + head * HEADD + c;
    *(bf16x8*)&sQ[row * ALD + c] = *(const bf16x8*)src;
  }
#endif

  v8f oacc[2][4];
#pragma unroll
  for (int i = 0; i < 2; ++i)
#pragma unroll
    for (int j = 0; j < 4; ++j)
#pragma unroll
      for (int v = 0; v < 8; ++v) oacc[i][j][v] = 0.0f;

  float m_i = -1e30f, l_i = 0.0f;

  for (int kt = 0; kt <= qb; ++kt) {
    const size_t tok0k = (size_t)b * SEQ + (size_t)kt * 128;
#ifdef HAVE_TDM
    if (tid == 0)    // TDM: K tile
      tdm_load_tile_128x128_bf16(lds_off(sK), Kmat + tok0k * (NKVH * HEADD) + kvh * HEADD,
                                 NKVH * HEADD);
#else
#pragma unroll
    for (int r = 0; r < 8; ++r) {
      int e = tid + r * 256;
      int row = e >> 4, c = (e & 15) * 8;
      const bf16* src = Kmat + (tok0k + row) * (NKVH * HEADD) + kvh * HEADD + c;
      *(bf16x8*)&sK[row * ALD + c] = *(const bf16x8*)src;
    }
#endif
#pragma unroll
    for (int r = 0; r < 8; ++r) {             // V tile transposed -> [hd][key]
      int e = tid + r * 256;
      int n = e >> 4, c = (e & 15) * 8;
      bf16x8 vv = *(const bf16x8*)(V + (tok0k + n) * (NKVH * HEADD) + kvh * HEADD + c);
#pragma unroll
      for (int jj = 0; jj < 8; ++jj) sV[(c + jj) * ALD + n] = vv[jj];
    }
#ifdef HAVE_TDM
    if (tid == 0) __builtin_amdgcn_s_wait_tensorcnt(0);   // covers Q (first iter) + K
#endif
    __syncthreads();

    v8f sacc[2][4];                            // S = Q K^T
#pragma unroll
    for (int i = 0; i < 2; ++i)
#pragma unroll
      for (int j = 0; j < 4; ++j)
#pragma unroll
        for (int v = 0; v < 8; ++v) sacc[i][j][v] = 0.0f;
#pragma unroll
    for (int ks = 0; ks < 4; ++ks) {
      v16bf af[2], bg[4];
#pragma unroll
      for (int i = 0; i < 2; ++i) af[i] = ld_afrag(sQ, wm * 32 + i * 16 + lrow, ALD, ks * 32, hs);
#pragma unroll
      for (int j = 0; j < 4; ++j) bg[j] = ld_bfrag(sK, wn * 64 + j * 16 + lrow, ALD, ks * 32, hs);
#pragma unroll
      for (int i = 0; i < 2; ++i)
#pragma unroll
        for (int j = 0; j < 4; ++j)
          sacc[i][j] = __builtin_amdgcn_wmma_f32_16x16x32_bf16(
              false, af[i], false, bg[j], (short)0, sacc[i][j], false, false);
    }
#pragma unroll
    for (int i = 0; i < 2; ++i)
#pragma unroll
      for (int j = 0; j < 4; ++j)
#pragma unroll
        for (int v = 0; v < 8; ++v) {
          int row = wm * 32 + i * 16 + v + 8 * hs;
          int col = wn * 64 + j * 16 + lrow;
          sS[row * SLD + col] = sacc[i][j][v] * scale;
        }
    __syncthreads();

    if (tid < 128) {                           // online softmax, one thread per row
      const int nvalid = (kt < qb) ? 128 : (tid + 1);
      const float* srow = &sS[tid * SLD];
      bf16* prow = &sP[tid * PLD];
      float rmax = -1e30f;
      for (int c = 0; c < nvalid; ++c) rmax = fmaxf(rmax, srow[c]);
      float newm = fmaxf(m_i, rmax);
      float sc = __expf(m_i - newm);
      float sum = 0.0f;
      for (int c = 0; c < 128; ++c) {
        float p = (c < nvalid) ? __expf(srow[c] - newm) : 0.0f;
        sum += p;
        prow[c] = f2bf(p);                     // overlays already-consumed f32 bytes
      }
      l_i = l_i * sc + sum;
      m_i = newm;
      sScale[tid] = sc;
    }
    __syncthreads();

#pragma unroll
    for (int i = 0; i < 2; ++i) {              // rescale O accumulators
      float scl[8];
#pragma unroll
      for (int v = 0; v < 8; ++v) scl[v] = sScale[wm * 32 + i * 16 + v + 8 * hs];
#pragma unroll
      for (int j = 0; j < 4; ++j)
#pragma unroll
        for (int v = 0; v < 8; ++v) oacc[i][j][v] *= scl[v];
    }
#pragma unroll
    for (int ks = 0; ks < 4; ++ks) {           // O += P @ V
      v16bf af[2], bg[4];
#pragma unroll
      for (int i = 0; i < 2; ++i) af[i] = ld_afrag(sP, wm * 32 + i * 16 + lrow, PLD, ks * 32, hs);
#pragma unroll
      for (int j = 0; j < 4; ++j) bg[j] = ld_bfrag(sV, wn * 64 + j * 16 + lrow, ALD, ks * 32, hs);
#pragma unroll
      for (int i = 0; i < 2; ++i)
#pragma unroll
        for (int j = 0; j < 4; ++j)
          oacc[i][j] = __builtin_amdgcn_wmma_f32_16x16x32_bf16(
              false, af[i], false, bg[j], (short)0, oacc[i][j], false, false);
    }
    __syncthreads();
  }

  if (tid < 128) sScale[tid] = (l_i > 0.0f) ? (1.0f / l_i) : 0.0f;
  __syncthreads();
#pragma unroll
  for (int i = 0; i < 2; ++i)
#pragma unroll
    for (int j = 0; j < 4; ++j)
#pragma unroll
      for (int v = 0; v < 8; ++v) {
        int row = wm * 32 + i * 16 + v + 8 * hs;
        int col = wn * 64 + j * 16 + lrow;
        float val = oacc[i][j][v] * sScale[row];
        O[(tok0q + row) * (NHEAD * HEADD) + head * HEADD + col] = f2bf(val);
      }
}

// =====================  Elementwise / norm kernels  =====================
__global__ __launch_bounds__(256) void rmsnorm_cast_kernel(
    const float* __restrict__ x, const float* __restrict__ w, bf16* __restrict__ out)
{
  __shared__ float red[8];
  const int row = blockIdx.x, tid = threadIdx.x;
  const float* xr = x + (size_t)row * HIDDEN;
  float vals[8]; float ss = 0.0f;
#pragma unroll
  for (int j = 0; j < 8; ++j) { float v = xr[tid + j * 256]; vals[j] = v; ss += v * v; }
#pragma unroll
  for (int m = 16; m > 0; m >>= 1) ss += __shfl_xor(ss, m, 32);
  if ((tid & 31) == 0) red[tid >> 5] = ss;
  __syncthreads();
  float tot = 0.0f;
#pragma unroll
  for (int k = 0; k < 8; ++k) tot += red[k];
  float r = rsqrtf(tot / (float)HIDDEN + 1e-6f);
  bf16* orow = out + (size_t)row * HIDDEN;
#pragma unroll
  for (int j = 0; j < 8; ++j) orow[tid + j * 256] = f2bf(vals[j] * r * w[tid + j * 256]);
}

__global__ __launch_bounds__(256) void add_rms_cast_kernel(
    const float* __restrict__ a, const float* __restrict__ bmat,
    const float* __restrict__ w, float* __restrict__ x2, bf16* __restrict__ h2)
{
  __shared__ float red[8];
  const int row = blockIdx.x, tid = threadIdx.x;
  const size_t base = (size_t)row * HIDDEN;
  float vals[8]; float ss = 0.0f;
#pragma unroll
  for (int j = 0; j < 8; ++j) {
    int c = tid + j * 256;
    float v = a[base + c] + bmat[base + c];
    vals[j] = v; ss += v * v;
    x2[base + c] = v;
  }
#pragma unroll
  for (int m = 16; m > 0; m >>= 1) ss += __shfl_xor(ss, m, 32);
  if ((tid & 31) == 0) red[tid >> 5] = ss;
  __syncthreads();
  float tot = 0.0f;
#pragma unroll
  for (int k = 0; k < 8; ++k) tot += red[k];
  float r = rsqrtf(tot / (float)HIDDEN + 1e-6f);
#pragma unroll
  for (int j = 0; j < 8; ++j) {
    int c = tid + j * 256;
    h2[base + c] = f2bf(vals[j] * r * w[c]);
  }
}

__global__ __launch_bounds__(128) void rope_norm_kernel(
    const float* __restrict__ in, const float* __restrict__ nw,
    const float* __restrict__ cs, const float* __restrict__ sn,
    bf16* __restrict__ out, int nheads)
{
  __shared__ float red[4];
  __shared__ float nv[128];
  const int t = blockIdx.x, h = blockIdx.y, i = threadIdx.x;
  const size_t base = ((size_t)t * nheads + h) * HEADD;
  float v = in[base + i];
  float ss = v * v;
#pragma unroll
  for (int m = 16; m > 0; m >>= 1) ss += __shfl_xor(ss, m, 32);
  if ((i & 31) == 0) red[i >> 5] = ss;
  __syncthreads();
  float tot = red[0] + red[1] + red[2] + red[3];
  float r = rsqrtf(tot / 128.0f + 1e-6f);
  float n = v * r * nw[i];
  nv[i] = n;
  __syncthreads();
  float rot = (i < 64) ? -nv[i + 64] : nv[i - 64];
  float o = n * cs[(size_t)t * HEADD + i] + rot * sn[(size_t)t * HEADD + i];
  out[base + i] = f2bf(o);
}

__global__ __launch_bounds__(256) void cast_f32_bf16_kernel(
    const float* __restrict__ in, bf16* __restrict__ out, int n)
{
  int i = blockIdx.x * 256 + threadIdx.x;
  if (i < n) out[i] = f2bf(in[i]);
}

__global__ __launch_bounds__(256) void swiglu_kernel(
    const bf16* __restrict__ g, const bf16* __restrict__ u, bf16* __restrict__ out, int n)
{
  int i = blockIdx.x * 256 + threadIdx.x;
  if (i < n) {
    float gv = bf2f(g[i]);
    float uv = bf2f(u[i]);
    float s = gv / (1.0f + __expf(-gv));
    out[i] = f2bf(s * uv);
  }
}

__global__ __launch_bounds__(256) void add_kernel(
    const float* __restrict__ a, const float* __restrict__ b, float* __restrict__ o, int n)
{
  int i = blockIdx.x * 256 + threadIdx.x;
  if (i < n) o[i] = a[i] + b[i];
}

// =====================  Host launcher  =====================
extern "C" void kernel_launch(void* const* d_in, const int* in_sizes, int n_in,
                              void* d_out, int out_size, void* d_ws, size_t ws_size,
                              hipStream_t stream)
{
  (void)in_sizes; (void)n_in; (void)out_size; (void)ws_size;
  const float* x    = (const float*)d_in[0];
  const float* cosb = (const float*)d_in[1];
  const float* sinb = (const float*)d_in[2];
  // d_in[3] = attn_mask (causal; applied analytically in-kernel)
  const float* wq  = (const float*)d_in[4];
  const float* bq  = (const float*)d_in[5];
  const float* wk  = (const float*)d_in[6];
  const float* bk  = (const float*)d_in[7];
  const float* wv  = (const float*)d_in[8];
  const float* bv  = (const float*)d_in[9];
  const float* wo  = (const float*)d_in[10];
  const float* qnw = (const float*)d_in[11];
  const float* knw = (const float*)d_in[12];
  const float* ln1 = (const float*)d_in[13];
  const float* ln2 = (const float*)d_in[14];
  const float* wg  = (const float*)d_in[15];
  const float* wu  = (const float*)d_in[16];
  const float* wd  = (const float*)d_in[17];

  char* ws = (char*)d_ws;
  const size_t MB = 1ull << 20;
  bf16*  hbf    = (bf16*)(ws + 0);         // 16MB  (reused as h2 bf16)
  float* qf     = (float*)(ws + 16 * MB);  // 32MB  (reused: o_proj out, mlp out)
  float* kf     = (float*)(ws + 48 * MB);  // 8MB
  float* vf     = (float*)(ws + 56 * MB);  // 8MB
  bf16*  qbf    = (bf16*)(ws + 64 * MB);   // 16MB
  bf16*  kbf    = (bf16*)(ws + 80 * MB);   // 4MB
  bf16*  vbf    = (bf16*)(ws + 84 * MB);   // 4MB
  bf16*  attnbf = (bf16*)(ws + 88 * MB);   // 16MB
  float* x2     = (float*)(ws + 104 * MB); // 32MB
  bf16*  gatebf = (bf16*)(ws + 136 * MB);  // 64MB (reused as act in-place)
  bf16*  upbf   = (bf16*)(ws + 200 * MB);  // 64MB  -> peak 264MB

  // 1. h = rms(x, ln1) -> bf16
  rmsnorm_cast_kernel<<<TOK, 256, 0, stream>>>(x, ln1, hbf);
  // 2. q/k/v projections
  gemm_bf16_kernel<false, true><<<dim3(TOK / 128, (NHEAD * HEADD) / 128), 256, 0, stream>>>(
      hbf, wq, bq, qf, TOK, NHEAD * HEADD, HIDDEN);
  gemm_bf16_kernel<false, true><<<dim3(TOK / 128, (NKVH * HEADD) / 128), 256, 0, stream>>>(
      hbf, wk, bk, kf, TOK, NKVH * HEADD, HIDDEN);
  gemm_bf16_kernel<false, true><<<dim3(TOK / 128, (NKVH * HEADD) / 128), 256, 0, stream>>>(
      hbf, wv, bv, vf, TOK, NKVH * HEADD, HIDDEN);
  // 3. per-head rmsnorm + rope (q, k); cast v
  rope_norm_kernel<<<dim3(TOK, NHEAD), 128, 0, stream>>>(qf, qnw, cosb, sinb, qbf, NHEAD);
  rope_norm_kernel<<<dim3(TOK, NKVH), 128, 0, stream>>>(kf, knw, cosb, sinb, kbf, NKVH);
  cast_f32_bf16_kernel<<<(TOK * NKVH * HEADD) / 256, 256, 0, stream>>>(vf, vbf, TOK * NKVH * HEADD);
  // 4. causal GQA flash attention
  attention_kernel<<<dim3(SEQ / 128, NHEAD, BATCH), 256, 0, stream>>>(qbf, kbf, vbf, attnbf);
  // 5. output projection
  gemm_bf16_kernel<false, false><<<dim3(TOK / 128, HIDDEN / 128), 256, 0, stream>>>(
      attnbf, wo, nullptr, qf, TOK, HIDDEN, NHEAD * HEADD);
  // 6. residual + rms(ln2)
  add_rms_cast_kernel<<<TOK, 256, 0, stream>>>(x, qf, ln2, x2, hbf);
  // 7. gate / up (bf16 outputs)
  gemm_bf16_kernel<true, false><<<dim3(TOK / 128, FFDIM / 128), 256, 0, stream>>>(
      hbf, wg, nullptr, gatebf, TOK, FFDIM, HIDDEN);
  gemm_bf16_kernel<true, false><<<dim3(TOK / 128, FFDIM / 128), 256, 0, stream>>>(
      hbf, wu, nullptr, upbf, TOK, FFDIM, HIDDEN);
  // 8. silu(gate)*up (in place over gate buffer)
  swiglu_kernel<<<(TOK * FFDIM) / 256, 256, 0, stream>>>(gatebf, upbf, gatebf, TOK * FFDIM);
  // 9. down projection
  gemm_bf16_kernel<false, false><<<dim3(TOK / 128, HIDDEN / 128), 256, 0, stream>>>(
      gatebf, wd, nullptr, qf, TOK, HIDDEN, FFDIM);
  // 10. final residual
  add_kernel<<<(TOK * HIDDEN) / 256, 256, 0, stream>>>(x2, qf, (float*)d_out, TOK * HIDDEN);
}
